// TopologyHead_16269336117488
// MI455X (gfx1250) — matrix operand order repaired and checked by reading, not compiled, not run on hardware
//
#include <hip/hip_runtime.h>

typedef __attribute__((ext_vector_type(16))) _Float16 v16h;
typedef __attribute__((ext_vector_type(8)))  float    v8f;

#define BB   64
#define NN   4096
#define HH   64
#define WW   64
#define LL   5
#define RR   256
#define EPSF 1e-8f

// ---------------------------------------------------------------------------
// Kernel 1: persistence image as a per-batch 64x4096x64 GEMM on the WMMA path.
// One workgroup per batch, 256 threads = 8 waves (wave32).
// Wave (rt, kh): row tile rt=wid&3 (rows 16*rt..16*rt+15), K-half kh=wid>>2.
// A[m,k] = pers_k * exp(-(y_m - pers_k/(1+eps))^2 / (2 sigma^2))  (16x32 f16)
// B[k,n] = exp(-(x_n - birth_k/(1+eps))^2 / (2 sigma^2))          (32x16 f16)
// Fragments are synthesized in-register per the CDNA5 VGPR layouts.
// ---------------------------------------------------------------------------
__launch_bounds__(256, 1)
__global__ void pimage_wmma_kernel(const float* __restrict__ pairs,
                                   const float* __restrict__ sigma_p,
                                   float* __restrict__ out)
{
    __shared__ float s_birth[NN];          // 16 KB
    __shared__ float s_pers [NN];          // 16 KB (also reused for max-reduce)
    __shared__ float s_img[2][HH * WW];    // 32 KB (two K-half partial planes)

    const int b    = blockIdx.x;
    const int tid  = threadIdx.x;
    const int lane = tid & 31;
    const int wid  = tid >> 5;
    const int hg   = lane >> 4;   // half-wave group (0: lanes 0-15, 1: 16-31)
    const int ln   = lane & 15;

    const float sigma  = sigma_p[0];
    const float inv2s2 = 1.0f / (2.0f * sigma * sigma);
    const float c2     = inv2s2 * 1.44269504088896340736f;  // log2(e) folded in
    const float inv1pe = 1.0f / (1.0f + EPSF);

    // ---- stage point data into LDS --------------------------------------
    const float* pb = pairs + (size_t)b * NN * 2;
    for (int i = tid; i < NN; i += 256) {
        float bi = pb[2 * i + 0];
        float de = pb[2 * i + 1];
        s_birth[i] = bi;
        s_pers[i]  = de - bi;
    }
    __syncthreads();

    const int   rt = wid & 3;           // row tile 0..3
    const int   kh = wid >> 2;          // K half 0..1
    const int   h0 = rt * 16;
    const float yh = (float)(h0 + ln) * (1.0f / 63.0f);   // grid_y for A row m=ln

    float xw0 = (float)( 0 + ln) * (1.0f / 63.0f);        // grid_x, col tiles
    float xw1 = (float)(16 + ln) * (1.0f / 63.0f);
    float xw2 = (float)(32 + ln) * (1.0f / 63.0f);
    float xw3 = (float)(48 + ln) * (1.0f / 63.0f);

    v8f acc0 = {}, acc1 = {}, acc2 = {}, acc3 = {};

    const int k_begin = kh * (NN / 2);
    const int k_end   = k_begin + (NN / 2);
    for (int kk = k_begin; kk < k_end; kk += 32) {
        // ---- A fragment (16x32 f16): lane half e -> K = hg*16 + e
        v16h a;
        #pragma unroll
        for (int e = 0; e < 16; ++e) {
            int   K  = hg * 16 + e;
            float pr = s_pers[kk + K];
            float pn = pr * inv1pe;
            float d  = yh - pn;
            float g  = __builtin_amdgcn_exp2f(-c2 * d * d) * pr;   // gy * weight
            a[e] = (_Float16)g;
        }
        // ---- B fragments (32x16 f16): lane half e -> K = hg*16 + e
        v16h b0, b1, b2, b3;
        #pragma unroll
        for (int e = 0; e < 16; ++e) {
            int   K  = hg * 16 + e;
            float bn = s_birth[kk + K] * inv1pe;
            float d0 = xw0 - bn;
            float d1 = xw1 - bn;
            float d2 = xw2 - bn;
            float d3 = xw3 - bn;
            b0[e] = (_Float16)__builtin_amdgcn_exp2f(-c2 * d0 * d0);
            b1[e] = (_Float16)__builtin_amdgcn_exp2f(-c2 * d1 * d1);
            b2[e] = (_Float16)__builtin_amdgcn_exp2f(-c2 * d2 * d2);
            b3[e] = (_Float16)__builtin_amdgcn_exp2f(-c2 * d3 * d3);
        }
        // ---- 4 WMMA issues: 16x64 strip, K += 32 ------------------------
        acc0 = __builtin_amdgcn_wmma_f32_16x16x32_f16(false, a, false, b0,
                                                      (short)0, acc0, false, false);
        acc1 = __builtin_amdgcn_wmma_f32_16x16x32_f16(false, a, false, b1,
                                                      (short)0, acc1, false, false);
        acc2 = __builtin_amdgcn_wmma_f32_16x16x32_f16(false, a, false, b2,
                                                      (short)0, acc2, false, false);
        acc3 = __builtin_amdgcn_wmma_f32_16x16x32_f16(false, a, false, b3,
                                                      (short)0, acc3, false, false);
    }

    // ---- spill accumulators to this wave's LDS plane --------------------
    // C/D layout: VGPR r -> M = r + hg*8, lane n = ln
    #pragma unroll
    for (int r = 0; r < 8; ++r) {
        int h = h0 + r + hg * 8;
        s_img[kh][h * WW +  0 + ln] = acc0[r];
        s_img[kh][h * WW + 16 + ln] = acc1[r];
        s_img[kh][h * WW + 32 + ln] = acc2[r];
        s_img[kh][h * WW + 48 + ln] = acc3[r];
    }
    __syncthreads();

    // ---- per-batch max reduction (reuse s_pers as scratch) --------------
    float m = 0.0f;  // image entries are non-negative
    for (int i = tid; i < HH * WW; i += 256)
        m = fmaxf(m, s_img[0][i] + s_img[1][i]);
    s_pers[tid] = m;
    __syncthreads();
    for (int s = 128; s > 0; s >>= 1) {
        if (tid < s) s_pers[tid] = fmaxf(s_pers[tid], s_pers[tid + s]);
        __syncthreads();
    }
    const float scale = 1.0f / (s_pers[0] + EPSF);

    float* ob = out + (size_t)b * HH * WW;
    for (int i = tid; i < HH * WW; i += 256)
        ob[i] = (s_img[0][i] + s_img[1][i]) * scale;
}

// ---------------------------------------------------------------------------
// Kernel 2: persistence landscapes. One workgroup per batch, thread r owns one
// landscape sample: branch-free top-5 (sorted max/min insertion chain) over
// the 4096 LDS-broadcast tent values.
// ---------------------------------------------------------------------------
__launch_bounds__(256, 1)
__global__ void plandscape_kernel(const float* __restrict__ pairs,
                                  float* __restrict__ out)
{
    __shared__ float s_b[NN];     // 16 KB
    __shared__ float s_d[NN];     // 16 KB
    __shared__ float s_mn[256];
    __shared__ float s_mx[256];

    const int b   = blockIdx.x;
    const int tid = threadIdx.x;

    const float* pb = pairs + (size_t)b * NN * 2;
    float mn = 3.4e38f, mx = -3.4e38f;
    for (int i = tid; i < NN; i += 256) {
        float bi = pb[2 * i + 0];
        float de = pb[2 * i + 1];
        s_b[i] = bi;
        s_d[i] = de;
        mn = fminf(mn, bi);
        mx = fmaxf(mx, de);
    }
    s_mn[tid] = mn;
    s_mx[tid] = mx;
    __syncthreads();
    for (int s = 128; s > 0; s >>= 1) {
        if (tid < s) {
            s_mn[tid] = fminf(s_mn[tid], s_mn[tid + s]);
            s_mx[tid] = fmaxf(s_mx[tid], s_mx[tid + s]);
        }
        __syncthreads();
    }
    const float minb = s_mn[0];
    const float maxd = s_mx[0];
    const float t = minb + (maxd - minb) * ((float)tid * (1.0f / 255.0f));

    // top-5 registers, descending; tents >= 0 so 0-init matches top_k padding
    float t0 = 0.f, t1 = 0.f, t2 = 0.f, t3 = 0.f, t4 = 0.f;
    for (int n = 0; n < NN; ++n) {
        float bi = s_b[n];
        float de = s_d[n];
        float x  = fmaxf(fminf(t - bi, de - t), 0.0f);    // tent value
        float u;
        u = fmaxf(t0, x); x = fminf(t0, x); t0 = u;
        u = fmaxf(t1, x); x = fminf(t1, x); t1 = u;
        u = fmaxf(t2, x); x = fminf(t2, x); t2 = u;
        u = fmaxf(t3, x); x = fminf(t3, x); t3 = u;
        t4 = fmaxf(t4, x);
    }

    float* ob = out + (size_t)b * LL * RR;
    ob[0 * RR + tid] = t0;
    ob[1 * RR + tid] = t1;
    ob[2 * RR + tid] = t2;
    ob[3 * RR + tid] = t3;
    ob[4 * RR + tid] = t4;
}

extern "C" void kernel_launch(void* const* d_in, const int* in_sizes, int n_in,
                              void* d_out, int out_size, void* d_ws, size_t ws_size,
                              hipStream_t stream) {
    const float* pairs = (const float*)d_in[0];
    const float* sigma = (const float*)d_in[1];
    float*       out   = (float*)d_out;

    pimage_wmma_kernel<<<dim3(BB), dim3(256), 0, stream>>>(pairs, sigma, out);
    plandscape_kernel <<<dim3(BB), dim3(256), 0, stream>>>(pairs, out + BB * HH * WW);
}
